// ChunkwiseGatedDeltaRule_91079076479237
// MI455X (gfx1250) — matrix operand order repaired and checked by reading, not compiled, not run on hardware
//
#include <hip/hip_runtime.h>

// ---------------------------------------------------------------------------
// Chunkwise Gated Delta Rule for MI455X (gfx1250, wave32, WMMA bf16)
//   B=16, T=4096, D=1024, C=128, chunk L=64
// Pipeline:
//   1) proj_kernel     : q/k/v/g = x @ W^T (+bias, k-L2-norm, g-sigmoid), bf16 out,
//                        WMMA with double-buffered LDS panels
//   2) ab_kernel       : alpha/beta = sigmoid(x @ Wa/Wb^T) (memory bound, VALU)
//   3) recurrence      : chunkwise delta rule, state M (128x128 f32) in LDS,
//                        TDM tensor_load_to_lds for K/Q/V tiles (TENSORcnt),
//                        async global->LDS for the gate tile (ASYNCcnt),
//                        next-chunk prefetch (global_prefetch_b8),
//                        all chunk matmuls via v_wmma_f32_16x16x32_bf16
//   4) outproj_kernel  : out = (o*g) @ Wo^T + bo, WMMA, f32 out
// ---------------------------------------------------------------------------

#define DEV __device__ __forceinline__

typedef __attribute__((ext_vector_type(16))) __bf16 bf16x16;
typedef __attribute__((ext_vector_type(8)))  float  f32x8;

constexpr int Bb = 16, Tt = 4096, Dd = 1024, Cc = 128;
constexpr int BT = Bb * Tt;     // 65536 rows
constexpr int L  = 64;          // chunk length
constexpr int NC = Tt / L;      // 64 chunks

DEV float sigm(float x) { return 1.0f / (1.0f + __expf(-x)); }

// ---- CDNA5 async global->LDS copy (ASYNCcnt path), guarded ----------------
#if defined(__has_builtin)
#if __has_builtin(__builtin_amdgcn_global_load_async_to_lds_b128) && \
    __has_builtin(__builtin_amdgcn_s_wait_asynccnt)
#define HAVE_ASYNC_LDS 1
#endif
#if __has_builtin(__builtin_amdgcn_tensor_load_to_lds) && \
    __has_builtin(__builtin_amdgcn_s_wait_tensorcnt)
#define HAVE_TDM 1
#endif
#endif
#ifndef HAVE_ASYNC_LDS
#define HAVE_ASYNC_LDS 0
#endif
#ifndef HAVE_TDM
#define HAVE_TDM 0
#endif

#if HAVE_ASYNC_LDS
// Builtin signature (from clang diagnostic): param0 = v4i addrspace(1)*,
// param1 = v4i addrspace(3)*, then two imm ints (offset, cpol).
typedef int v4i_ __attribute__((__vector_size__(16)));
typedef __attribute__((address_space(1))) v4i_ GV4;
typedef __attribute__((address_space(3))) v4i_ LV4;
// generic LDS pointer: ADDR[63:32]=shared aperture, ADDR[31:0]=LDS offset
// (ISA 10.2 aperture rules), so the low 32 bits are the AS(3) value.
DEV void async_copy16(void* lds, const void* g) {
  __builtin_amdgcn_global_load_async_to_lds_b128(
      (GV4*)(unsigned long long)g,
      (LV4*)(unsigned)(unsigned long long)lds, 0, 0);
}
DEV void async_wait0() { __builtin_amdgcn_s_wait_asynccnt(0); }
#else
DEV void async_copy16(void* lds, const void* g) {
  *(uint4*)lds = *(const uint4*)g;
}
DEV void async_wait0() {}
#endif

// ---- CDNA5 Tensor Data Mover: 2D bf16 tile, global -> LDS -----------------
#if HAVE_TDM
typedef unsigned int u32x4_ __attribute__((ext_vector_type(4)));
typedef int          i32x4_ __attribute__((ext_vector_type(4)));
typedef int          i32x8_ __attribute__((ext_vector_type(8)));
// D# per ISA 8.3/8.4: group0 = {count/type, lds_addr, global_addr[56:0]},
// group1 = {data_size, tensor_dim0/1, tile_dim0/1, dim0_stride}; groups 2/3
// zero (2D tensor, no dims 2..4). This toolchain uses the 6-arg builtin:
// (g0, g1, g2, g3, extra int32x8, cpol). Issued once per wave (TENSORcnt).
DEV void tdm_load_tile_bf16(void* lds, const void* g,
                            int rows, int cols, int stride_elems) {
  const unsigned long long ga = (unsigned long long)g;
  u32x4_ g0;
  g0[0] = 1u;                                    // count=1 (valid), user mode
  g0[1] = (unsigned)(unsigned long long)lds;     // lds_addr (low 32 = AS3)
  g0[2] = (unsigned)(ga & 0xffffffffu);          // global_addr[31:0]
  g0[3] = (unsigned)((ga >> 32) & 0x01ffffffu)   // global_addr[56:32]
        | (2u << 30);                            // type = 2 ("image")
  i32x8_ g1;
  g1[0] = (1 << 16);                             // data_size=1 (2B), mask=0
  g1[1] = (cols & 0xffff) << 16;                 // tensor_dim0[15:0]
  g1[2] = ((cols >> 16) & 0xffff)                // tensor_dim0[31:16]
        | ((rows & 0xffff) << 16);               // tensor_dim1[15:0]
  g1[3] = ((rows >> 16) & 0xffff)                // tensor_dim1[31:16]
        | ((cols & 0xffff) << 16);               // tile_dim0
  g1[4] = rows & 0xffff;                         // tile_dim1 (tile_dim2=0)
  g1[5] = stride_elems;                          // tensor_dim0_stride[31:0]
  g1[6] = 0;                                     // stride hi, dim1_stride lo
  g1[7] = 0;                                     // dim1_stride hi
  i32x4_ z4 = {0, 0, 0, 0};
  i32x8_ z8 = {0, 0, 0, 0, 0, 0, 0, 0};
  __builtin_amdgcn_tensor_load_to_lds(g0, g1, z4, z4, z8, 0);
}
DEV void tdm_wait0() { __builtin_amdgcn_s_wait_tensorcnt(0); }
#else
DEV void tdm_wait0() {}
#endif

// ---- WMMA fragment loaders (wave32; contiguous k order, same in A and B) ----------
// A (16x32 bf16): lane l -> row m=l&15; elems e = K (l>>4)*16+e (contiguous)
DEV bf16x16 ldA(const __bf16* base, int ld) {
  const int l = threadIdx.x & 31, hi = l >> 4, m = l & 15;
  const __bf16* p = base + m * ld + hi * 16;
  bf16x16 f;
#pragma unroll
  for (int e = 0; e < 16; ++e) f[e] = p[e];
  return f;
}
// B (32x16 bf16): lane l -> col n=l&15; elems e = K (l>>4)*16+e (contiguous).
// baseT must point at storage laid out as Bt[n][kk] (i.e. B transposed).
DEV bf16x16 ldB(const __bf16* baseT, int ld) {
  const int l = threadIdx.x & 31, hi = l >> 4, n = l & 15;
  const __bf16* p = baseT + n * ld + hi * 16;
  bf16x16 f;
#pragma unroll
  for (int e = 0; e < 16; ++e) f[e] = p[e];
  return f;
}
DEV f32x8 wmma(bf16x16 a, bf16x16 b, f32x8 c) {
  return __builtin_amdgcn_wmma_f32_16x16x32_bf16(false, a, false, b,
                                                 (short)0, c, false, false);
}
// D element r of lane l maps to row (r + (l>>4)*8), col (l&15).

// ===========================================================================
// Kernel 1: fused q/k/v/g projection.  grid (BT/128, 4), block 256 (8 waves).
// 128x128 tile; K loop over D in steps of 32 with double-buffered panels.
// ===========================================================================
__global__ __launch_bounds__(256) void proj_kernel(
    const float* __restrict__ x,
    const float* __restrict__ Wq, const float* __restrict__ bq,
    const float* __restrict__ Wk, const float* __restrict__ bk,
    const float* __restrict__ Wv, const float* __restrict__ bv,
    const float* __restrict__ Wg, const float* __restrict__ bg,
    __bf16* __restrict__ qb, __bf16* __restrict__ kb,
    __bf16* __restrict__ vb, __bf16* __restrict__ gb)
{
  __shared__ __bf16 Ap[2][128 * 32];  // x tile   (rows x K-slice), bf16
  __shared__ __bf16 Bp[2][128 * 32];  // W tile   (out-col x K-slice), bf16
  __shared__ float  rss[128];         // row sum-of-squares for k-normalize

  const int which = blockIdx.y;
  const float* W    = which == 0 ? Wq : which == 1 ? Wk : which == 2 ? Wv : Wg;
  const float* bias = which == 0 ? bq : which == 1 ? bk : which == 2 ? bv : bg;
  __bf16* out       = which == 0 ? qb : which == 1 ? kb : which == 2 ? vb : gb;

  const int r0  = blockIdx.x * 128;
  const int tid = threadIdx.x;
  const int w = tid >> 5, rw = w >> 1, cw = w & 1;   // 4x2 wave grid
  const int l = tid & 31, hi = l >> 4, n = l & 15;

  auto load_panel = [&](int buf, int kk) {
    for (int i = tid; i < 1024; i += 256) {          // 128 rows x 8 float4
      const int row = i >> 3, c4 = (i & 7) * 4;
      float4 xv = *(const float4*)(x + (size_t)(r0 + row) * Dd + kk + c4);
      __bf16* da = Ap[buf] + row * 32 + c4;
      da[0] = (__bf16)xv.x; da[1] = (__bf16)xv.y;
      da[2] = (__bf16)xv.z; da[3] = (__bf16)xv.w;
      float4 wv = *(const float4*)(W + (size_t)row * Dd + kk + c4);
      __bf16* db = Bp[buf] + row * 32 + c4;
      db[0] = (__bf16)wv.x; db[1] = (__bf16)wv.y;
      db[2] = (__bf16)wv.z; db[3] = (__bf16)wv.w;
    }
  };

  f32x8 acc[2][4];
#pragma unroll
  for (int a2 = 0; a2 < 2; ++a2)
#pragma unroll
    for (int b2 = 0; b2 < 4; ++b2)
#pragma unroll
      for (int e = 0; e < 8; ++e) acc[a2][b2][e] = 0.0f;

  load_panel(0, 0);
  int p = 0;
  for (int kk = 0; kk < Dd; kk += 32) {
    __syncthreads();                       // panel p ready
    if (kk + 32 < Dd) load_panel(p ^ 1, kk + 32);   // overlap with compute
    bf16x16 afr[2];
#pragma unroll
    for (int a2 = 0; a2 < 2; ++a2)
      afr[a2] = ldA(Ap[p] + (rw * 32 + a2 * 16) * 32, 32);
#pragma unroll
    for (int b2 = 0; b2 < 4; ++b2) {
      bf16x16 bfr = ldB(Bp[p] + (cw * 64 + b2 * 16) * 32, 32);
#pragma unroll
      for (int a2 = 0; a2 < 2; ++a2) acc[a2][b2] = wmma(afr[a2], bfr, acc[a2][b2]);
    }
    p ^= 1;
  }

  // ---- epilogue -----------------------------------------------------------
  if (which == 1) {                 // k: L2-normalize each row over C=128
    if (tid < 128) rss[tid] = 0.0f;
    __syncthreads();
#pragma unroll
    for (int a2 = 0; a2 < 2; ++a2)
#pragma unroll
      for (int b2 = 0; b2 < 4; ++b2)
#pragma unroll
        for (int r = 0; r < 8; ++r) {
          const int row_l = rw * 32 + a2 * 16 + r + hi * 8;
          const int col   = cw * 64 + b2 * 16 + n;
          const float v = acc[a2][b2][r] + bias[col];
          atomicAdd(&rss[row_l], v * v);
        }
    __syncthreads();
#pragma unroll
    for (int a2 = 0; a2 < 2; ++a2)
#pragma unroll
      for (int b2 = 0; b2 < 4; ++b2)
#pragma unroll
        for (int r = 0; r < 8; ++r) {
          const int row_l = rw * 32 + a2 * 16 + r + hi * 8;
          const int col   = cw * 64 + b2 * 16 + n;
          const float v = acc[a2][b2][r] + bias[col];
          const float s = 1.0f / fmaxf(sqrtf(rss[row_l]), 1e-12f);
          out[(size_t)(r0 + row_l) * Cc + col] = (__bf16)(v * s);
        }
  } else {
#pragma unroll
    for (int a2 = 0; a2 < 2; ++a2)
#pragma unroll
      for (int b2 = 0; b2 < 4; ++b2)
#pragma unroll
        for (int r = 0; r < 8; ++r) {
          const int row_l = rw * 32 + a2 * 16 + r + hi * 8;
          const int col   = cw * 64 + b2 * 16 + n;
          float v = acc[a2][b2][r] + bias[col];
          if (which == 3) v = sigm(v);               // gate
          out[(size_t)(r0 + row_l) * Cc + col] = (__bf16)v;
        }
  }
}

// ===========================================================================
// Kernel 2: alpha/beta scalar gates. One wave per row, memory bound.
// ===========================================================================
__global__ __launch_bounds__(256) void ab_kernel(
    const float* __restrict__ x,
    const float* __restrict__ Wa, const float* __restrict__ ba,
    const float* __restrict__ Wb, const float* __restrict__ bb2,
    float* __restrict__ af, float* __restrict__ bfv)
{
  const int r = blockIdx.x * 8 + (threadIdx.x >> 5);
  const int lane = threadIdx.x & 31;
  const float* xr = x + (size_t)r * Dd;
  float sa = 0.0f, sb = 0.0f;
#pragma unroll
  for (int it = 0; it < 8; ++it) {
    const int c = it * 128 + lane * 4;
    float4 xv = *(const float4*)(xr + c);
    float4 wa = *(const float4*)(Wa + c);
    float4 wb = *(const float4*)(Wb + c);
    sa += xv.x * wa.x + xv.y * wa.y + xv.z * wa.z + xv.w * wa.w;
    sb += xv.x * wb.x + xv.y * wb.y + xv.z * wb.z + xv.w * wb.w;
  }
#pragma unroll
  for (int off = 16; off > 0; off >>= 1) {
    sa += __shfl_xor(sa, off, 32);
    sb += __shfl_xor(sb, off, 32);
  }
  if (lane == 0) { af[r] = sigm(sa + ba[0]); bfv[r] = sigm(sb + bb2[0]); }
}

// ===========================================================================
// Kernel 3: chunkwise gated delta rule. grid = B (16 persistent blocks),
// block = 256 threads (8 waves). All per-chunk matmuls are bf16 WMMA.
//
// Recurrence (M = S^T, key x value):
//   M_t = a_t (I - b_t k_t k_t^T) M_{t-1} + b_t k_t v_t^T ,  o_t = M_t q_t
// Chunk (L=64, gamma_t = prod a):
//   (I + T) U = diag(b) V - diag(b*gamma) (K M0),
//     T_{ts} = b_t (g_t/g_s)(k_t.k_s),  s<t   (UT transform, unit lower tri)
//   O  = diag(gamma) (Q M0^T) + tril((g_t/g_s) Q U^T) K
//   M1 = gamma_L M0 + K^T diag(gamma_L/gamma_t) U
// ===========================================================================
__global__ __launch_bounds__(256) void recurrence_kernel(
    const __bf16* __restrict__ qb, const __bf16* __restrict__ kb,
    const __bf16* __restrict__ vb, const __bf16* __restrict__ gb,
    const float* __restrict__ af, const float* __restrict__ bfv,
    __bf16* __restrict__ og)
{
  extern __shared__ char smem_raw[];
  float*  Mf   = (float*)smem_raw;              // 128x128 f32 state
  __bf16* Mrow = (__bf16*)(Mf + 128 * 128);     // 128x128 bf16 M (row major)
  __bf16* Mcol = Mrow + 128 * 128;              // 128x128 bf16 M^T
  __bf16* Kc   = Mcol + 128 * 128;              // 64x128 keys
  __bf16* KcT  = Kc + 64 * 128;                 // 128x64 keys^T
  __bf16* Qc   = KcT + 128 * 64;                // 64x128 queries
  __bf16* VG   = Qc + 64 * 128;                 // 64x128 values, then gate
  __bf16* Ub   = VG + 64 * 128;                 // 64x128 U (bf16)
  __bf16* UbT  = Ub + 64 * 128;                 // 128x64 U^T (bf16)
  __bf16* U2T  = UbT + 128 * 64;                // 128x64 (gL/g_t)U^T (bf16)
  __bf16* Tmb  = U2T + 128 * 64;                // 64x64 block-strict-lower T
  __bf16* Amb  = Tmb + 64 * 64;                 // 64x64 attention scores
  float*  Uc   = (float*)(Amb + 64 * 64);       // 64x128 U (f32)
  float*  Tm   = Uc + 64 * 128;                 // 64x64 strict-lower T (f32)
  float*  avec = Tm + 64 * 64;
  float*  bvec = avec + 64;
  float*  gam  = bvec + 64;

  const int tid = threadIdx.x, w = tid >> 5;
  const int l = tid & 31, hi = l >> 4, n = l & 15;
  const int batch = blockIdx.x;

  for (int i = tid; i < 128 * 128; i += 256) {
    Mf[i] = 0.0f; Mrow[i] = (__bf16)0.0f; Mcol[i] = (__bf16)0.0f;
  }
  for (int i = tid; i < 64 * 128; i += 256) {
    Ub[i] = (__bf16)0.0f; UbT[i] = (__bf16)0.0f; U2T[i] = (__bf16)0.0f;
  }
  __syncthreads();

  for (int ch = 0; ch < NC; ++ch) {
    const size_t rbase = (size_t)batch * Tt + (size_t)ch * L;

    // (1) decay scalars + cumulative product gamma
    if (tid < 64) { avec[tid] = af[rbase + tid]; bvec[tid] = bfv[rbase + tid]; }
    __syncthreads();
    if (tid == 0) {
      float g = 1.0f;
      for (int t = 0; t < 64; ++t) { g *= avec[t]; gam[t] = g; }
    }
    __syncthreads();
    const float gamL = gam[63];

    // (2) stage K/Q/V chunk tiles (64x128 bf16 each)
#if HAVE_TDM
    // one TDM descriptor per tensor, issued by waves 0/1/2 (TENSORcnt)
    if (w == 0)      tdm_load_tile_bf16(Kc, kb + rbase * (size_t)Cc, L, Cc, Cc);
    else if (w == 1) tdm_load_tile_bf16(Qc, qb + rbase * (size_t)Cc, L, Cc, Cc);
    else if (w == 2) tdm_load_tile_bf16(VG, vb + rbase * (size_t)Cc, L, Cc, Cc);
#else
    for (int i = tid; i < 64 * 16; i += 256) {
      const int row = i >> 4, c8 = (i & 15) * 8;
      const size_t src = (rbase + row) * (size_t)Cc + c8;
      async_copy16(Kc + row * 128 + c8, kb + src);
      async_copy16(Qc + row * 128 + c8, qb + src);
      async_copy16(VG + row * 128 + c8, vb + src);
    }
#endif
    // prefetch next chunk (64 KB) into cache while this chunk computes
    if (ch + 1 < NC) {
      const size_t nb = (rbase + L) * (size_t)Cc + (size_t)tid * 32;
      __builtin_prefetch(kb + nb, 0, 0);
      __builtin_prefetch(qb + nb, 0, 0);
      __builtin_prefetch(vb + nb, 0, 0);
      __builtin_prefetch(gb + nb, 0, 0);
    }
    tdm_wait0();
    async_wait0();
    __syncthreads();
    for (int i = tid; i < 128 * 64; i += 256) {
      const int j = i >> 6, t = i & 63;
      KcT[j * 64 + t] = Kc[t * 128 + j];
    }
    __syncthreads();

    // (3) T = tril(b_t (g_t/g_s) K K^T), 64x64, kdim=128 ----------------
    for (int f = 0; f < 2; ++f) {
      const int fid = w * 2 + f, fr = fid >> 2, fc = fid & 3;
      f32x8 acc = {};
      for (int kk = 0; kk < 128; kk += 32)
        acc = wmma(ldA(Kc + fr * 16 * 128 + kk, 128),
                   ldB(Kc + fc * 16 * 128 + kk, 128), acc);
#pragma unroll
      for (int r = 0; r < 8; ++r) {
        const int t = fr * 16 + r + hi * 8, s = fc * 16 + n;
        const float tv = (s < t) ? bvec[t] * (gam[t] / gam[s]) * acc[r] : 0.0f;
        Tm[t * 64 + s]  = tv;
        Tmb[t * 64 + s] = (__bf16)((s < (t & ~15)) ? tv : 0.0f);
      }
    }
    __syncthreads();

    // (4) RHS: Uc = diag(b)V - diag(b*gamma) (K @ M0), 64x128, kdim=128 --
    for (int f = 0; f < 4; ++f) {
      const int fid = w * 4 + f, fr = fid >> 3, fc = fid & 7;
      f32x8 acc = {};
      for (int kk = 0; kk < 128; kk += 32)
        acc = wmma(ldA(Kc + fr * 16 * 128 + kk, 128),
                   ldB(Mcol + fc * 16 * 128 + kk, 128), acc);
#pragma unroll
      for (int r = 0; r < 8; ++r) {
        const int t = fr * 16 + r + hi * 8, i2 = fc * 16 + n;
        const float vv = (float)VG[t * 128 + i2];
        Uc[t * 128 + i2] = bvec[t] * vv - bvec[t] * gam[t] * acc[r];
      }
    }
    __syncthreads();

    // stage gate chunk into VG via async global->LDS (ASYNCcnt path);
    // overlapped with the solve, waited on just before matmul (8).
    for (int i = tid; i < 64 * 16; i += 256) {
      const int row = i >> 4, c8 = (i & 15) * 8;
      async_copy16(VG + row * 128 + c8,
                   gb + (rbase + row) * (size_t)Cc + c8);
    }

    // (5) blocked unit-lower-triangular solve (I+T) U = RHS --------------
    for (int bi = 0; bi < 4; ++bi) {
      if (bi > 0) {   // off-diagonal update via WMMA (Tmb is 0 above block)
        f32x8 acc = {};
        for (int kk = 0; kk < 64; kk += 32)
          acc = wmma(ldA(Tmb + bi * 16 * 64 + kk, 64),
                     ldB(UbT + w * 16 * 64 + kk, 64), acc);
#pragma unroll
        for (int r = 0; r < 8; ++r) {
          const int t = bi * 16 + r + hi * 8, i2 = w * 16 + n;
          Uc[t * 128 + i2] -= acc[r];
        }
        __syncthreads();
      }
      // 16x16 diagonal solve, fully in registers: lane owns one column
      if (l < 16) {
        const int i2 = w * 16 + l;
        float u[16];
#pragma unroll
        for (int tt = 0; tt < 16; ++tt) {
          float xv = Uc[(bi * 16 + tt) * 128 + i2];
          for (int ss = 0; ss < tt; ++ss)
            xv -= Tm[(bi * 16 + tt) * 64 + bi * 16 + ss] * u[ss];
          u[tt] = xv;
        }
#pragma unroll
        for (int tt = 0; tt < 16; ++tt) {
          const int t = bi * 16 + tt;
          Uc[t * 128 + i2] = u[tt];
          const __bf16 ub = (__bf16)u[tt];
          Ub[t * 128 + i2] = ub;
          UbT[i2 * 64 + t] = ub;
        }
      }
      __syncthreads();
    }

    // (6) U2T = (gamma_L / gamma_t) U^T (bf16) ---------------------------
    for (int i = tid; i < 128 * 64; i += 256) {
      const int col = i >> 6, t = i & 63;
      U2T[col * 64 + t] = (__bf16)((gamL / gam[t]) * Uc[t * 128 + col]);
    }
    __syncthreads();

    // (7) Amb = tril((g_t/g_s) Q U^T), 64x64, kdim=128 -------------------
    for (int f = 0; f < 2; ++f) {
      const int fid = w * 2 + f, fr = fid >> 2, fc = fid & 3;
      f32x8 acc = {};
      for (int kk = 0; kk < 128; kk += 32)
        acc = wmma(ldA(Qc + fr * 16 * 128 + kk, 128),
                   ldB(Ub + fc * 16 * 128 + kk, 128), acc);
#pragma unroll
      for (int r = 0; r < 8; ++r) {
        const int t = fr * 16 + r + hi * 8, s = fc * 16 + n;
        Amb[t * 64 + s] = (__bf16)((s <= t) ? (gam[t] / gam[s]) * acc[r] : 0.0f);
      }
    }
    async_wait0();       // gate tile (async) complete before (8) reads it
    __syncthreads();

    // (8) O = diag(gamma)(Q M0^T) + Amb @ K ; gate + store (bf16) --------
    for (int f = 0; f < 4; ++f) {
      const int fid = w * 4 + f, fr = fid >> 3, fc = fid & 7;
      f32x8 acc = {};
      for (int kk = 0; kk < 128; kk += 32)
        acc = wmma(ldA(Qc + fr * 16 * 128 + kk, 128),
                   ldB(Mrow + fc * 16 * 128 + kk, 128), acc);
#pragma unroll
      for (int r = 0; r < 8; ++r) acc[r] *= gam[fr * 16 + r + hi * 8];
      for (int kk = 0; kk < 64; kk += 32)
        acc = wmma(ldA(Amb + fr * 16 * 64 + kk, 64),
                   ldB(KcT + fc * 16 * 64 + kk, 64), acc);
#pragma unroll
      for (int r = 0; r < 8; ++r) {
        const int t = fr * 16 + r + hi * 8, j = fc * 16 + n;
        const float ov = acc[r] * (float)VG[t * 128 + j];   // apply gate
        og[(rbase + t) * (size_t)Cc + j] = (__bf16)ov;
      }
    }
    __syncthreads();

    // (9) state: M = gamma_L*M + K^T @ (diag(gL/g_t)U), 128x128, kdim=64 -
    for (int f = 0; f < 8; ++f) {
      const int fid = w * 8 + f, fr = fid >> 3, fc = fid & 7;
      f32x8 acc;
#pragma unroll
      for (int r = 0; r < 8; ++r)
        acc[r] = gamL * Mf[(fr * 16 + r + hi * 8) * 128 + fc * 16 + n];
      for (int kk = 0; kk < 64; kk += 32)
        acc = wmma(ldA(KcT + fr * 16 * 64 + kk, 64),
                   ldB(U2T + fc * 16 * 64 + kk, 64), acc);
#pragma unroll
      for (int r = 0; r < 8; ++r) {
        const int j = fr * 16 + r + hi * 8, i2 = fc * 16 + n;
        const float mv = acc[r];
        Mf[j * 128 + i2]   = mv;
        Mrow[j * 128 + i2] = (__bf16)mv;
        Mcol[i2 * 128 + j] = (__bf16)mv;
      }
    }
    __syncthreads();
  }
}

constexpr size_t SMEM_BYTES =
    (size_t)128 * 128 * 4        // Mf
  + (size_t)2 * 128 * 128 * 2    // Mrow, Mcol
  + (size_t)7 * 64 * 128 * 2     // Kc, KcT, Qc, VG, Ub, UbT, U2T
  + (size_t)2 * 64 * 64 * 2      // Tmb, Amb
  + (size_t)64 * 128 * 4         // Uc
  + (size_t)64 * 64 * 4          // Tm
  + (size_t)3 * 64 * 4;          // avec, bvec, gam   (= 312064 B < 320 KB)

// ===========================================================================
// Kernel 4: out = og @ Wo^T + bo.  grid (BT/128, D/128), block 256.
// ===========================================================================
__global__ __launch_bounds__(256) void outproj_kernel(
    const __bf16* __restrict__ og, const float* __restrict__ Wo,
    const float* __restrict__ bo, float* __restrict__ out)
{
  __shared__ __bf16 Ap[2][128 * 32];
  __shared__ __bf16 Bp[2][128 * 32];
  const int r0 = blockIdx.x * 128, c0 = blockIdx.y * 128;
  const int tid = threadIdx.x;
  const int w = tid >> 5, rw = w >> 1, cw = w & 1;
  const int l = tid & 31, hi = l >> 4, n = l & 15;

  auto load_panel = [&](int buf, int kk) {
    for (int i = tid; i < 512; i += 256) {           // og is already bf16
      const int row = i >> 2, c8 = (i & 3) * 8;
      *(uint4*)(Ap[buf] + row * 32 + c8) =
          *(const uint4*)(og + (size_t)(r0 + row) * Cc + kk + c8);
    }
    for (int i = tid; i < 1024; i += 256) {          // Wo f32 -> bf16
      const int row = i >> 3, c4 = (i & 7) * 4;
      float4 wv = *(const float4*)(Wo + (size_t)(c0 + row) * Cc + kk + c4);
      __bf16* db = Bp[buf] + row * 32 + c4;
      db[0] = (__bf16)wv.x; db[1] = (__bf16)wv.y;
      db[2] = (__bf16)wv.z; db[3] = (__bf16)wv.w;
    }
  };

  f32x8 acc[2][4];
#pragma unroll
  for (int a2 = 0; a2 < 2; ++a2)
#pragma unroll
    for (int b2 = 0; b2 < 4; ++b2)
#pragma unroll
      for (int e = 0; e < 8; ++e) acc[a2][b2][e] = 0.0f;

  load_panel(0, 0);
  int p = 0;
  for (int kk = 0; kk < Cc; kk += 32) {
    __syncthreads();
    if (kk + 32 < Cc) load_panel(p ^ 1, kk + 32);
    bf16x16 afr[2];
#pragma unroll
    for (int a2 = 0; a2 < 2; ++a2)
      afr[a2] = ldA(Ap[p] + (rw * 32 + a2 * 16) * 32, 32);
#pragma unroll
    for (int b2 = 0; b2 < 4; ++b2) {
      bf16x16 bfr = ldB(Bp[p] + (cw * 64 + b2 * 16) * 32, 32);
#pragma unroll
      for (int a2 = 0; a2 < 2; ++a2) acc[a2][b2] = wmma(afr[a2], bfr, acc[a2][b2]);
    }
    p ^= 1;
  }
#pragma unroll
  for (int a2 = 0; a2 < 2; ++a2)
#pragma unroll
    for (int b2 = 0; b2 < 4; ++b2)
#pragma unroll
      for (int r = 0; r < 8; ++r) {
        const int row_l = rw * 32 + a2 * 16 + r + hi * 8;
        const int col   = cw * 64 + b2 * 16 + n;
        out[(size_t)(r0 + row_l) * Dd + c0 + col] = acc[a2][b2][r] + bo[c0 + col];
      }
}

// ===========================================================================
extern "C" void kernel_launch(void* const* d_in, const int* in_sizes, int n_in,
                              void* d_out, int out_size, void* d_ws, size_t ws_size,
                              hipStream_t stream) {
  (void)in_sizes; (void)n_in; (void)out_size; (void)ws_size;
  const float* x   = (const float*)d_in[0];
  const float* Wq  = (const float*)d_in[1];
  const float* bq  = (const float*)d_in[2];
  const float* Wk  = (const float*)d_in[3];
  const float* bk  = (const float*)d_in[4];
  const float* Wv  = (const float*)d_in[5];
  const float* bv  = (const float*)d_in[6];
  const float* Wa  = (const float*)d_in[7];
  const float* ba  = (const float*)d_in[8];
  const float* Wb  = (const float*)d_in[9];
  const float* bb2 = (const float*)d_in[10];
  const float* Wg  = (const float*)d_in[11];
  const float* bg  = (const float*)d_in[12];
  const float* Wo  = (const float*)d_in[13];
  const float* bo  = (const float*)d_in[14];
  float* out = (float*)d_out;

  __bf16* qb = (__bf16*)d_ws;
  __bf16* kb = qb + (size_t)BT * Cc;
  __bf16* vb = kb + (size_t)BT * Cc;
  __bf16* gb = vb + (size_t)BT * Cc;
  __bf16* og = gb + (size_t)BT * Cc;
  float*  af  = (float*)(og + (size_t)BT * Cc);
  float*  bfv = af + BT;

  proj_kernel<<<dim3(BT / 128, 4), 256, 0, stream>>>(
      x, Wq, bq, Wk, bk, Wv, bv, Wg, bg, qb, kb, vb, gb);
  ab_kernel<<<dim3(BT / 8), 256, 0, stream>>>(x, Wa, ba, Wb, bb2, af, bfv);

  (void)hipFuncSetAttribute((const void*)recurrence_kernel,
                            hipFuncAttributeMaxDynamicSharedMemorySize,
                            (int)SMEM_BYTES);
  recurrence_kernel<<<dim3(Bb), 256, SMEM_BYTES, stream>>>(
      qb, kb, vb, gb, af, bfv, og);

  outproj_kernel<<<dim3(BT / 128, Dd / 128), 256, 0, stream>>>(og, Wo, bo, out);
}
